// SelectiveSSM_4767413698792
// MI455X (gfx1250) — compile-verified, matrix operand order checked by reading, not run on hardware
//
#include <hip/hip_runtime.h>
#include <hip/hip_bf16.h>
#include <math.h>
#include <stdint.h>

typedef __attribute__((ext_vector_type(16))) _Float16 v16h;
typedef __attribute__((ext_vector_type(8)))  _Float16 v8h;
typedef __attribute__((ext_vector_type(8)))  float    v8f;

#define B_SZ   2
#define L_SZ   4096
#define DMODEL 1024
#define DSTATE 64
#define DCONV  4
#define DINNER 2048
#define NROWS  (B_SZ * L_SZ)            /* 8192 token rows */
#define DBC_N  (2 * DSTATE + DINNER)    /* 2176 */
#define EPSF   1.1920929e-07f

#define TM 128   /* block tile rows (8 waves x 16) */
#define TN 64    /* block tile cols (4 wmma subtiles) */
#define TK 32    /* K step = wmma K */

// ---------------------------------------------------------------------------
// LDS-staged WMMA GEMM: C[M,N] = A[M,K] @ B[K,N] (+ bias[N]).
//  - A tile (128x32 f32) staged via global_load_async_to_lds_b128 (ASYNCcnt)
//  - B tile (32x64 f32) staged via float4 loads, transposed+converted to f16
//  - fragments read from LDS as aligned b128 chunks; f32 accum in 4x v8f
// Block = 256 threads (8 waves). Grid = (N/64, M/128). Dims are tile-multiples.
// ---------------------------------------------------------------------------
__global__ void __launch_bounds__(256) gemm_f16wmma(
    const float* __restrict__ A, const float* __restrict__ Bm,
    float* __restrict__ C, const float* __restrict__ bias,
    int K, int lda, int ldb, int ldc) {
  __shared__ float    sA[TM][TK];       // raw f32 A tile (async-staged)
  __shared__ _Float16 sB[TN][TK + 8];   // f16 B tile, transposed [n][k], padded

  const int tid  = threadIdx.x;
  const int lane = tid & 31;
  const int wave = tid >> 5;
  const int m_blk = blockIdx.y * TM;
  const int n_blk = blockIdx.x * TN;

  const int kb   = (lane & 16) ? 8 : 0;      // K-half select per ISA layout
  const int mrow = wave * 16 + (lane & 15);  // A-frag row within block tile
  const int ncol = lane & 15;                // B/C-frag col within subtile

  v8f acc[4] = {v8f{}, v8f{}, v8f{}, v8f{}};

  for (int k0 = 0; k0 < K; k0 += TK) {
    __syncthreads();  // protect LDS tiles from previous iteration's readers

    // ---- stage A tile: 1024 float4 chunks, 4 per thread, async to LDS ----
#pragma unroll
    for (int i = 0; i < 4; ++i) {
      const int lin = tid + i * 256;      // 0..1023
      const int row = lin >> 3;           // 8 float4 per row
      const int kq  = (lin & 7) << 2;
      const float* gp = A + (size_t)(m_blk + row) * lda + (k0 + kq);
      const unsigned ldsa = (unsigned)(uintptr_t)&sA[row][kq];
      asm volatile("global_load_async_to_lds_b128 %0, %1, off"
                   :: "v"(ldsa), "v"((unsigned long long)(uintptr_t)gp)
                   : "memory");
    }

    // ---- stage B tile: 512 float4 chunks, 2 per thread, transpose->f16 ----
#pragma unroll
    for (int i = 0; i < 2; ++i) {
      const int lin = tid + i * 256;      // 0..511
      const int kr  = lin >> 4;           // 16 float4 per k-row
      const int nq  = (lin & 15) << 2;
      const float4 v = *(const float4*)(Bm + (size_t)(k0 + kr) * ldb + (n_blk + nq));
      sB[nq + 0][kr] = (_Float16)v.x;
      sB[nq + 1][kr] = (_Float16)v.y;
      sB[nq + 2][kr] = (_Float16)v.z;
      sB[nq + 3][kr] = (_Float16)v.w;
    }
    if (k0 + TK < K)  // prefetch next B tile into L2 (global_prefetch_b8)
      __builtin_prefetch(Bm + (size_t)(k0 + TK) * ldb + n_blk + (tid & 31) * 4, 0, 1);

    asm volatile("s_wait_asynccnt 0x0" ::: "memory");  // drain async A copies
    __syncthreads();

    // ---- build A fragment (ISA 16-bit A 16x32 layout) from LDS ----
    const float* pa = &sA[mrow][kb];
    const float4 f0 = *(const float4*)(pa + 0);
    const float4 f1 = *(const float4*)(pa + 4);
    const float4 f2 = *(const float4*)(pa + 16);
    const float4 f3 = *(const float4*)(pa + 20);
    v16h a;
    a[0]  = (_Float16)f0.x; a[1]  = (_Float16)f0.y;
    a[2]  = (_Float16)f0.z; a[3]  = (_Float16)f0.w;
    a[4]  = (_Float16)f1.x; a[5]  = (_Float16)f1.y;
    a[6]  = (_Float16)f1.z; a[7]  = (_Float16)f1.w;
    a[8]  = (_Float16)f2.x; a[9]  = (_Float16)f2.y;
    a[10] = (_Float16)f2.z; a[11] = (_Float16)f2.w;
    a[12] = (_Float16)f3.x; a[13] = (_Float16)f3.y;
    a[14] = (_Float16)f3.z; a[15] = (_Float16)f3.w;

    // ---- 4 B fragments + 4 WMMA ----
#pragma unroll
    for (int j = 0; j < 4; ++j) {
      const int col = j * 16 + ncol;
      const v8h lo = *(const v8h*)&sB[col][kb];
      const v8h hi = *(const v8h*)&sB[col][kb + 16];
      const v16h b = __builtin_shufflevector(lo, hi,
          0, 1, 2, 3, 4, 5, 6, 7, 8, 9, 10, 11, 12, 13, 14, 15);
      acc[j] = __builtin_amdgcn_wmma_f32_16x16x32_f16(
          false, a, false, b, (short)0, acc[j], false, false);
    }
  }

  // C/D layout: VGPR r, lane -> row = m0 + r + 8*(lane>=16), col = n0+j*16+(lane&15)
  const int rbase = m_blk + wave * 16 + ((lane & 16) ? 8 : 0);
  const int c0 = n_blk + ncol;
#pragma unroll
  for (int j = 0; j < 4; ++j) {
    const int col = c0 + j * 16;
    const float bv = bias ? bias[col] : 0.0f;
#pragma unroll
    for (int r = 0; r < 8; ++r) {
      C[(size_t)(rbase + r) * ldc + col] = acc[j][r] + bv;
    }
  }
}

// ---------------------------------------------------------------------------
// Depthwise causal conv (width 4) + bias + SiLU over x_ssm = xz[..., :DINNER]
// ---------------------------------------------------------------------------
__global__ void conv_silu_kernel(const float* __restrict__ xz,
                                 const float* __restrict__ conv_w,
                                 const float* __restrict__ conv_b,
                                 float* __restrict__ xc) {
  const int idx = blockIdx.x * blockDim.x + threadIdx.x;   // over NROWS*DINNER
  const int d   = idx & (DINNER - 1);
  const int row = idx >> 11;          // DINNER = 2^11
  const int l   = row & (L_SZ - 1);
  const int b   = row >> 12;          // L = 2^12
  float s = conv_b[d];
#pragma unroll
  for (int j = 0; j < DCONV; ++j) {
    const int ll = l - (DCONV - 1) + j;
    if (ll >= 0)
      s += xz[(size_t)(b * L_SZ + ll) * (2 * DINNER) + d] * conv_w[d * DCONV + j];
  }
  xc[idx] = s / (1.0f + __expf(-s));   // SiLU
}

// ---------------------------------------------------------------------------
// Selective scan: one thread per (batch, channel); h[64] and exp(A_log)[64]
// live in VGPRs; B_t/C_t broadcast through LDS each timestep.
// Grid = 16 blocks x 256 threads (2 batches * 2048 channels).
// ---------------------------------------------------------------------------
__global__ void scan_kernel(const float* __restrict__ xc, const float* __restrict__ dt_in,
                            const float* __restrict__ dbc, const float* __restrict__ A_log,
                            const float* __restrict__ D_param, float* __restrict__ y) {
  const int tid = threadIdx.x;
  const int b   = blockIdx.x >> 3;                 // 8 blocks per batch
  const int d   = ((blockIdx.x & 7) << 8) + tid;   // channel index
  __shared__ float sB[DSTATE];
  __shared__ float sC[DSTATE];

  float eA[DSTATE], h[DSTATE];
#pragma unroll
  for (int s = 0; s < DSTATE; ++s) {
    eA[s] = __expf(A_log[(size_t)d * DSTATE + s]);   // A = -exp(A_log)
    h[s]  = 0.0f;
  }
  const float Dp = D_param[d];

  for (int l = 0; l < L_SZ; ++l) {
    const size_t row = (size_t)b * L_SZ + l;
    __syncthreads();                                 // WAR on sB/sC
    if (tid < 2 * DSTATE) {
      const float v = dbc[row * DBC_N + tid];
      if (tid < DSTATE) sB[tid] = v; else sC[tid - DSTATE] = v;
    }
    __syncthreads();

    const float dtr = dt_in[row * DINNER + d];
    const float dtv = (dtr > 20.0f) ? dtr : log1pf(__expf(dtr));  // softplus
    const float xv  = xc[row * DINNER + d];
    const float dx  = dtv * xv;
    float acc = 0.0f;
#pragma unroll
    for (int s = 0; s < DSTATE; ++s) {
      const float dA = __expf(-dtv * eA[s]);
      h[s] = fmaf(dA, h[s], dx * sB[s]);
      acc  = fmaf(h[s], sC[s], acc);
    }
    y[row * DINNER + d] = fmaf(Dp, xv, acc);
  }
}

// ---------------------------------------------------------------------------
// RMSNorm over DINNER + gate by SiLU(z); one block per token row.
// ---------------------------------------------------------------------------
__global__ void rmsnorm_gate_kernel(const float* __restrict__ y, const float* __restrict__ xz,
                                    const float* __restrict__ norm_w, float* __restrict__ g) {
  const int row = blockIdx.x;
  const int tid = threadIdx.x;
  __shared__ float red[256];
  float ss = 0.0f;
  for (int d = tid; d < DINNER; d += 256) {
    const float v = y[(size_t)row * DINNER + d];
    ss = fmaf(v, v, ss);
  }
  red[tid] = ss;
  __syncthreads();
  for (int off = 128; off > 0; off >>= 1) {
    if (tid < off) red[tid] += red[tid + off];
    __syncthreads();
  }
  const float rs = rsqrtf(red[0] * (1.0f / (float)DINNER) + EPSF);
  for (int d = tid; d < DINNER; d += 256) {
    const float v = y[(size_t)row * DINNER + d] * rs * norm_w[d];
    const float z = xz[(size_t)row * (2 * DINNER) + DINNER + d];
    const float sz = z / (1.0f + __expf(-z));
    g[(size_t)row * DINNER + d] = v * sz;
  }
}

// ---------------------------------------------------------------------------
// Host-side orchestration
// ---------------------------------------------------------------------------
extern "C" void kernel_launch(void* const* d_in, const int* in_sizes, int n_in,
                              void* d_out, int out_size, void* d_ws, size_t ws_size,
                              hipStream_t stream) {
  const float* x       = (const float*)d_in[0];
  const float* W_in    = (const float*)d_in[1];
  const float* conv_w  = (const float*)d_in[2];
  const float* conv_b  = (const float*)d_in[3];
  const float* W_x     = (const float*)d_in[4];
  const float* W_dt    = (const float*)d_in[5];
  const float* b_dt    = (const float*)d_in[6];
  const float* A_log   = (const float*)d_in[7];
  const float* D_param = (const float*)d_in[8];
  const float* W_out   = (const float*)d_in[9];
  const float* norm_w  = (const float*)d_in[10];
  float* out = (float*)d_out;

  char* ws = (char*)d_ws;
  float* xz  = (float*)(ws);                                   // 8192*4096 f32 (134.2 MB)
  float* xc  = (float*)(ws + (size_t)134217728);               // 8192*2048 f32 ( 67.1 MB)
  float* dbc = (float*)(ws + (size_t)201326592);               // 8192*2176 f32 ( 71.3 MB)
  float* dtb = (float*)(ws + (size_t)272629760);               // 8192*2048 f32 ( 67.1 MB)
  float* yb  = (float*)(ws + (size_t)339738624);               // 8192*2048 f32 ( 67.1 MB)
  float* gb  = dtb;  // dt is dead after the scan; reuse for gated activations

  const dim3 blk(256);

  // 1) xz = x @ W_in                      [8192,1024]x[1024,4096]
  gemm_f16wmma<<<dim3((2 * DINNER) / TN, NROWS / TM), blk, 0, stream>>>(
      x, W_in, xz, nullptr, DMODEL, DMODEL, 2 * DINNER, 2 * DINNER);

  // 2) xc = SiLU(causal_conv(xz[...,:DINNER]) + conv_b)
  conv_silu_kernel<<<(NROWS * DINNER) / 256, blk, 0, stream>>>(xz, conv_w, conv_b, xc);

  // 3) dbc = xc @ W_x                     [8192,2048]x[2048,2176]
  gemm_f16wmma<<<dim3(DBC_N / TN, NROWS / TM), blk, 0, stream>>>(
      xc, W_x, dbc, nullptr, DINNER, DINNER, DBC_N, DBC_N);

  // 4) dt = dbc[:,128:] @ W_dt + b_dt     [8192,2048]x[2048,2048]
  gemm_f16wmma<<<dim3(DINNER / TN, NROWS / TM), blk, 0, stream>>>(
      dbc + 2 * DSTATE, W_dt, dtb, b_dt, DINNER, DBC_N, DINNER, DINNER);

  // 5) selective scan -> yb
  scan_kernel<<<(B_SZ * DINNER) / 256, blk, 0, stream>>>(xc, dtb, dbc, A_log, D_param, yb);

  // 6) RMSNorm + SiLU(z) gating -> gb (overwrites dt buffer)
  rmsnorm_gate_kernel<<<NROWS, blk, 0, stream>>>(yb, xz, norm_w, gb);

  // 7) out = gb @ W_out                   [8192,2048]x[2048,1024]
  gemm_f16wmma<<<dim3(DMODEL / TN, NROWS / TM), blk, 0, stream>>>(
      gb, W_out, out, nullptr, DINNER, DINNER, DMODEL, DMODEL);
}
